// MSA_53979148976474
// MI455X (gfx1250) — matrix operand order repaired and checked by reading, not compiled, not run on hardware
//
#include <hip/hip_runtime.h>
#include <hip/hip_bf16.h>

typedef __attribute__((ext_vector_type(16))) __bf16 v16bf;
typedef __attribute__((ext_vector_type(8)))  float  v8f;

#define NWIN  4096
#define NTOK  64
#define DIM   192
#define HEADS 6
#define HD    32
#define C3    576
#define SCALE 0.17677669529663687f   // 1/sqrt(32)

__device__ __forceinline__ unsigned short f2bf(float f) {
  union { float f; unsigned u; } x; x.f = f;
  unsigned r = x.u + 0x7FFFu + ((x.u >> 16) & 1u);  // round-to-nearest-even
  return (unsigned short)(r >> 16);
}
__device__ __forceinline__ unsigned pk2bf(float lo, float hi) {
  return (unsigned)f2bf(lo) | ((unsigned)f2bf(hi) << 16);
}

union FragB16 { v16bf v; uint4 q[2]; };

// ---------------- prelude: swizzle a [OUT x IN] fp32 weight into WMMA
// B-fragment order for B = W^T (IN x OUT). B(k,n): lane = n + 16*(k>>4),
// half = k & 15. Fragment f = ntile*(IN/32) + kstep, 512 bf16 per fragment.
__global__ void swizzleB(const float* __restrict__ W, unsigned short* __restrict__ out,
                         int ntiles, int IN) {
  int tid = blockIdx.x * 256 + threadIdx.x;
  int ksteps = IN / 32;
  int total = ntiles * ksteps * 512;
  if (tid >= total) return;
  int half = tid & 15;
  int lane = (tid >> 4) & 31;
  int fi   = tid >> 9;
  int ks = fi % ksteps;
  int nt = fi / ksteps;
  int n_l = lane & 15;
  int k_l = half + 16 * (lane >> 4);
  out[tid] = f2bf(W[(nt * 16 + n_l) * IN + ks * 32 + k_l]);
}

// ---------------- prelude: expand relative-position bias to [H][64][64] f32
__global__ void expand_bias(const float* __restrict__ bt, const int* __restrict__ ri,
                            float* __restrict__ out) {
  int tid = blockIdx.x * 256 + threadIdx.x;
  if (tid >= HEADS * NTOK * NTOK) return;
  int h = tid / (NTOK * NTOK);
  int rem = tid - h * (NTOK * NTOK);
  out[tid] = bt[ri[rem] * HEADS + h];
}

// ---------------- main fused window-attention kernel: one workgroup per window
__launch_bounds__(256)
__global__ void win_attn(const float* __restrict__ x,
                         const float* __restrict__ qkv_b,
                         const float* __restrict__ proj_b,
                         const unsigned short* __restrict__ wqkvB,
                         const unsigned short* __restrict__ wprojB,
                         const float* __restrict__ biasE,
                         float* __restrict__ out) {
  __shared__ unsigned short lds_qk[NTOK][392];      // Q (0..191, pre-scaled) | K (192..383)
  __shared__ unsigned short lds_xo[NTOK][200];      // X (stage A), then O (stage C/D)
  __shared__ unsigned short lds_vt[DIM][72];        // V^T [channel][token]
  __shared__ unsigned short lds_p[HEADS][16][72];   // per-head P staging (16-row block)

  const int win  = blockIdx.x;
  const int tid  = threadIdx.x;
  const int lane = tid & 31;
  const int wave = tid >> 5;
  const int lhi  = lane >> 4;    // 0/1
  const int llo  = lane & 15;

  // ---- load X tile (64x192 fp32), convert to bf16 in LDS (packed b64 stores) ----
  {
    const float* xg = x + (size_t)win * (NTOK * DIM);
    int row = tid >> 2;                 // 4 threads per row
    int cb  = (tid & 3) * 48;
    const float4* src = (const float4*)(xg + row * DIM + cb);
#pragma unroll
    for (int i = 0; i < 12; ++i) {
      float4 f = src[i];
      uint2 p; p.x = pk2bf(f.x, f.y); p.y = pk2bf(f.z, f.w);
      *(uint2*)&lds_xo[row][cb + i * 4] = p;
    }
  }
  __syncthreads();

  // ---- stage A: QKV = X @ Wqkv^T + b  (scale folded into Q; V stored transposed) ----
  {
    const int mtile = wave & 3;
    const int ntb   = (wave >> 2) * 18;
    FragB16 a[6];
    const int tok = mtile * 16 + llo;
#pragma unroll
    for (int ks = 0; ks < 6; ++ks) {   // A(m,k): lane=m+16*(k>>4), half=k&15
      a[ks].q[0] = *(const uint4*)&lds_xo[tok][ks * 32 + 16 * lhi];
      a[ks].q[1] = *(const uint4*)&lds_xo[tok][ks * 32 + 16 * lhi + 8];
    }
    for (int nt = ntb; nt < ntb + 18; ++nt) {
      v8f acc = {};
#pragma unroll
      for (int ks = 0; ks < 6; ++ks) {
        FragB16 b;
        const uint4* bp = (const uint4*)(wqkvB + (((nt * 6 + ks) * 32) + lane) * 16);
        b.q[0] = bp[0]; b.q[1] = bp[1];
        acc = __builtin_amdgcn_wmma_f32_16x16x32_bf16(false, a[ks].v, false, b.v,
                                                      (short)0, acc, false, false);
      }
      const int o = nt * 16 + llo;
      const float bo = qkv_b[o];
      // region select is uniform per nt-tile (boundaries are multiples of 16):
      // branch on scalar nt so the compiler emits s_cbranch, not exec masking.
      if (nt < 12) {               // Q, pre-scaled
#pragma unroll
        for (int r = 0; r < 8; ++r)
          lds_qk[mtile * 16 + r + 8 * lhi][o] = f2bf((acc[r] + bo) * SCALE);
      } else if (nt < 24) {        // K
#pragma unroll
        for (int r = 0; r < 8; ++r)
          lds_qk[mtile * 16 + r + 8 * lhi][o] = f2bf(acc[r] + bo);
      } else {                     // V -> transposed
#pragma unroll
        for (int r = 0; r < 8; ++r)
          lds_vt[o - 384][mtile * 16 + r + 8 * lhi] = f2bf(acc[r] + bo);
      }
    }
  }
  __syncthreads();

  // ---- stages B+C: per-head attention (waves 0..5 own one head each) ----
  if (wave < HEADS) {
    const int h = wave;
    const int qbase = h * HD;
    const int kbase = 192 + h * HD;
    const float* bh = biasE + h * (NTOK * NTOK);

    FragB16 kb[4];                       // K^T B-frags: B(k=ch,n=kv-token)
#pragma unroll
    for (int nt = 0; nt < 4; ++nt) {
      const uint4* p = (const uint4*)&lds_qk[nt * 16 + llo][kbase + 16 * lhi];
      kb[nt].q[0] = p[0]; kb[nt].q[1] = p[1];
    }
    FragB16 vb[2][2];                    // V B-frags: B(k=kv-token,n=ch)
#pragma unroll
    for (int ntile = 0; ntile < 2; ++ntile)
#pragma unroll
      for (int kk = 0; kk < 2; ++kk) {
        const uint4* p = (const uint4*)&lds_vt[h * HD + ntile * 16 + llo][kk * 32 + 16 * lhi];
        vb[ntile][kk].q[0] = p[0]; vb[ntile][kk].q[1] = p[1];
      }

    for (int mt = 0; mt < 4; ++mt) {
      FragB16 qa;                        // Q A-frag (16 tokens x 32 ch)
      {
        const int tq = mt * 16 + llo;
        qa.q[0] = *(const uint4*)&lds_qk[tq][qbase + 16 * lhi];
        qa.q[1] = *(const uint4*)&lds_qk[tq][qbase + 16 * lhi + 8];
      }
      v8f s[4];
#pragma unroll
      for (int nt = 0; nt < 4; ++nt) {
        v8f z = {};
        s[nt] = __builtin_amdgcn_wmma_f32_16x16x32_bf16(false, qa.v, false, kb[nt].v,
                                                        (short)0, z, false, false);
      }
#pragma unroll
      for (int nt = 0; nt < 4; ++nt)
#pragma unroll
        for (int r = 0; r < 8; ++r) {
          const int q = mt * 16 + r + 8 * lhi;
          s[nt][r] += bh[q * 64 + nt * 16 + llo];
        }
      // row softmax: 16 columns live across lanes of a half, 4 tiles cover kv 0..63
      float inv[8];
#pragma unroll
      for (int r = 0; r < 8; ++r) {
        float m = fmaxf(fmaxf(s[0][r], s[1][r]), fmaxf(s[2][r], s[3][r]));
        m = fmaxf(m, __shfl_xor(m, 1, 32));
        m = fmaxf(m, __shfl_xor(m, 2, 32));
        m = fmaxf(m, __shfl_xor(m, 4, 32));
        m = fmaxf(m, __shfl_xor(m, 8, 32));
        float sum = 0.f;
#pragma unroll
        for (int nt = 0; nt < 4; ++nt) { s[nt][r] = __expf(s[nt][r] - m); sum += s[nt][r]; }
        sum += __shfl_xor(sum, 1, 32);
        sum += __shfl_xor(sum, 2, 32);
        sum += __shfl_xor(sum, 4, 32);
        sum += __shfl_xor(sum, 8, 32);
        inv[r] = __builtin_amdgcn_rcpf(sum);
      }
#pragma unroll
      for (int nt = 0; nt < 4; ++nt)
#pragma unroll
        for (int r = 0; r < 8; ++r)
          lds_p[h][r + 8 * lhi][nt * 16 + llo] = f2bf(s[nt][r] * inv[r]);

      // stage C: O block = P(16x64) @ V(64x32)
      FragB16 pa[2];
#pragma unroll
      for (int kk = 0; kk < 2; ++kk) {
        pa[kk].q[0] = *(const uint4*)&lds_p[h][llo][kk * 32 + 16 * lhi];
        pa[kk].q[1] = *(const uint4*)&lds_p[h][llo][kk * 32 + 16 * lhi + 8];
      }
#pragma unroll
      for (int ntile = 0; ntile < 2; ++ntile) {
        v8f oacc = {};
        oacc = __builtin_amdgcn_wmma_f32_16x16x32_bf16(false, pa[0].v, false, vb[ntile][0].v,
                                                       (short)0, oacc, false, false);
        oacc = __builtin_amdgcn_wmma_f32_16x16x32_bf16(false, pa[1].v, false, vb[ntile][1].v,
                                                       (short)0, oacc, false, false);
#pragma unroll
        for (int r = 0; r < 8; ++r)
          lds_xo[mt * 16 + r + 8 * lhi][h * HD + ntile * 16 + llo] = f2bf(oacc[r]);
      }
    }
  }
  __syncthreads();

  // ---- stage D: Y = O @ Wproj^T + b, fp32 store ----
  {
    const int mtile = wave & 3;
    const int ntb   = (wave >> 2) * 6;
    FragB16 a[6];
    const int tok = mtile * 16 + llo;
#pragma unroll
    for (int ks = 0; ks < 6; ++ks) {
      a[ks].q[0] = *(const uint4*)&lds_xo[tok][ks * 32 + 16 * lhi];
      a[ks].q[1] = *(const uint4*)&lds_xo[tok][ks * 32 + 16 * lhi + 8];
    }
    float* og = out + (size_t)win * (NTOK * DIM);
    for (int nt = ntb; nt < ntb + 6; ++nt) {
      v8f acc = {};
#pragma unroll
      for (int ks = 0; ks < 6; ++ks) {
        FragB16 b;
        const uint4* bp = (const uint4*)(wprojB + (((nt * 6 + ks) * 32) + lane) * 16);
        b.q[0] = bp[0]; b.q[1] = bp[1];
        acc = __builtin_amdgcn_wmma_f32_16x16x32_bf16(false, a[ks].v, false, b.v,
                                                      (short)0, acc, false, false);
      }
      const int o = nt * 16 + llo;
      const float pb = proj_b[o];
#pragma unroll
      for (int r = 0; r < 8; ++r)
        og[(mtile * 16 + r + 8 * lhi) * DIM + o] = acc[r] + pb;
    }
  }
}

extern "C" void kernel_launch(void* const* d_in, const int* in_sizes, int n_in,
                              void* d_out, int out_size, void* d_ws, size_t ws_size,
                              hipStream_t stream) {
  const float* x      = (const float*)d_in[0];
  const float* qkv_w  = (const float*)d_in[1];
  const float* qkv_b  = (const float*)d_in[2];
  const float* proj_w = (const float*)d_in[3];
  const float* proj_b = (const float*)d_in[4];
  const float* bias_t = (const float*)d_in[5];
  const int*   rel_i  = (const int*)d_in[6];
  (void)in_sizes; (void)n_in; (void)out_size; (void)ws_size;

  // workspace layout: [qkv B-frags bf16][proj B-frags bf16][bias f32]
  unsigned short* wqkvB  = (unsigned short*)d_ws;           // 36*6*512 halves
  unsigned short* wprojB = wqkvB + 36 * 6 * 512;            // 12*6*512 halves
  float*          biasE  = (float*)(wprojB + 12 * 6 * 512); // 6*64*64 floats

  swizzleB<<<(36 * 6 * 512 + 255) / 256, 256, 0, stream>>>(qkv_w, wqkvB, 36, DIM);
  swizzleB<<<(12 * 6 * 512 + 255) / 256, 256, 0, stream>>>(proj_w, wprojB, 12, DIM);
  expand_bias<<<(HEADS * NTOK * NTOK + 255) / 256, 256, 0, stream>>>(bias_t, rel_i, biasE);

  win_attn<<<NWIN, 256, 0, stream>>>(x, qkv_b, proj_b, wqkvB, wprojB, biasE,
                                     (float*)d_out);
}
